// CrossAttention_3100966388338
// MI455X (gfx1250) — matrix operand order repaired and checked by reading, not compiled, not run on hardware
//
#include <hip/hip_runtime.h>
#include <hip/hip_bf16.h>

// ---------------------------------------------------------------------------
// CDNA5 (gfx1250) cross-attention, bf16 WMMA pipeline.
//  - all matmuls via v_wmma_f32_16x16x32_bf16 (wave32)
//  - flash attention stages K/V tiles in LDS via global_load_async_to_lds_b128
//    (ASYNCcnt double-buffering), softmax reductions via DPP16 (no LDS)
// ---------------------------------------------------------------------------

typedef __attribute__((ext_vector_type(16))) __bf16 v16bf;
typedef __attribute__((ext_vector_type(8)))  __bf16 v8bf;
typedef __attribute__((ext_vector_type(4)))  __bf16 v4bf;
typedef __attribute__((ext_vector_type(8)))  float  v8f;

#define DIMK   1024
#define NHEADS 16
#define HDIM   64
#define BATCH  4
#define LQ     4096
#define LKV    1024
#define QSCALE 0.125f   // HEAD_DIM^-0.5

// ---- WMMA fragment helpers (this file's contiguous k order, same for A and B; 16-bit, wave32) ----
// A: 16x32 (MxK). lane<16: row=lane, K=k0..k0+15
//                 lane>=16: row=lane-16, K=k0+16..k0+31
__device__ __forceinline__ v16bf load_a_frag(const __bf16* base, int ld) {
  const int lane = threadIdx.x & 31;
  const int row  = lane & 15;
  const int koff = (lane & 16) ? 16 : 0;
  const __bf16* p = base + (size_t)row * ld + koff;
  v8bf lo = *(const v8bf*)p;
  v8bf hi = *(const v8bf*)(p + 8);
  v16bf r;
#pragma unroll
  for (int i = 0; i < 8; ++i) { r[i] = lo[i]; r[i + 8] = hi[i]; }
  return r;
}

// B: 32x16 (KxN), column n of B contiguous in memory (row n of (N,K) matrix).
// lane<16: col=lane, K=k0..k0+15 ; lane>=16: col=lane-16, K=k0+16..k0+31
__device__ __forceinline__ v16bf load_b_frag(const __bf16* base, int ld) {
  const int lane = threadIdx.x & 31;
  const int col  = lane & 15;
  const int koff = (lane & 16) ? 16 : 0;
  return *(const v16bf*)(base + (size_t)col * ld + koff);
}

__device__ __forceinline__ v8f wmma_bf16(v16bf a, v16bf b, v8f c) {
  return __builtin_amdgcn_wmma_f32_16x16x32_bf16(false, a, false, b,
                                                 (short)0, c, false, false);
}

// ---- DPP16 16-lane butterfly reductions (stay within each 16-lane half) ----
// steps: quad_perm xor1 (0xB1), quad_perm xor2 (0x4E),
//        ROW_HALF_MIRROR (0x141, valid xor4-combine once quads uniform),
//        ROW_MIRROR (0x140, valid xor8-combine once groups-of-8 uniform)
template <int CTRL>
__device__ __forceinline__ float dpp_xchg(float x) {
  int y = __builtin_amdgcn_update_dpp(0, __builtin_bit_cast(int, x),
                                      CTRL, 0xF, 0xF, true);
  return __builtin_bit_cast(float, y);
}
__device__ __forceinline__ float red_max16(float x) {
  x = fmaxf(x, dpp_xchg<0xB1>(x));
  x = fmaxf(x, dpp_xchg<0x4E>(x));
  x = fmaxf(x, dpp_xchg<0x141>(x));
  x = fmaxf(x, dpp_xchg<0x140>(x));
  return x;
}
__device__ __forceinline__ float red_sum16(float x) {
  x += dpp_xchg<0xB1>(x);
  x += dpp_xchg<0x4E>(x);
  x += dpp_xchg<0x141>(x);
  x += dpp_xchg<0x140>(x);
  return x;
}

// ---- async global -> LDS staging (CDNA5, tracked by ASYNCcnt) ----
__device__ __forceinline__ void async_load_b128(uint32_t lds_off, const void* gptr) {
  asm volatile("global_load_async_to_lds_b128 %0, %1, off"
               :: "v"(lds_off), "v"((unsigned long long)(uintptr_t)gptr)
               : "memory");
}
__device__ __forceinline__ void wait_async_le2() {
  asm volatile("s_wait_asynccnt 0x2" ::: "memory");
}
__device__ __forceinline__ void wait_async_0() {
  asm volatile("s_wait_asynccnt 0x0" ::: "memory");
}

// ---------------------------------------------------------------------------
// fp32 -> bf16 cast (vectorized x4)
// ---------------------------------------------------------------------------
__global__ void cast_f32_to_bf16(const float* __restrict__ in,
                                 __bf16* __restrict__ out, int n4) {
  int i = blockIdx.x * blockDim.x + threadIdx.x;
  if (i >= n4) return;
  float4 f = ((const float4*)in)[i];
  v4bf o;
  o[0] = (__bf16)f.x; o[1] = (__bf16)f.y; o[2] = (__bf16)f.z; o[3] = (__bf16)f.w;
  ((v4bf*)out)[i] = o;
}

// ---------------------------------------------------------------------------
// GEMM: C[M,N] = (A[M,K] @ W[N,K]^T + bias) * scale
// block = 256 threads = 8 waves as 2(M) x 4(N); each wave a 32x32 register
// tile (2x2 WMMA accumulators -> 4 independent WMMA chains per k-step).
// ---------------------------------------------------------------------------
template <int OUT_F32>
__global__ void gemm_xWt(const __bf16* __restrict__ A, const __bf16* __restrict__ W,
                         const float* __restrict__ bias, void* __restrict__ Cout,
                         int M, int N, int K, float scale) {
  const int wave = threadIdx.x >> 5;
  const int lane = threadIdx.x & 31;
  const int mTile = blockIdx.x * 64 + (wave >> 2) * 32;
  const int nTile = blockIdx.y * 128 + (wave & 3) * 32;
  const __bf16* aBase0 = A + (size_t)mTile * K;
  const __bf16* aBase1 = aBase0 + (size_t)16 * K;
  const __bf16* wBase0 = W + (size_t)nTile * K;
  const __bf16* wBase1 = wBase0 + (size_t)16 * K;
  v8f acc00 = {}, acc01 = {}, acc10 = {}, acc11 = {};
  for (int k0 = 0; k0 < K; k0 += 32) {
    __builtin_prefetch(aBase0 + k0 + 64, 0, 1);   // global_prefetch_b8
    __builtin_prefetch(wBase0 + k0 + 64, 0, 1);
    v16bf a0 = load_a_frag(aBase0 + k0, K);
    v16bf a1 = load_a_frag(aBase1 + k0, K);
    v16bf b0 = load_b_frag(wBase0 + k0, K);
    v16bf b1 = load_b_frag(wBase1 + k0, K);
    acc00 = wmma_bf16(a0, b0, acc00);
    acc01 = wmma_bf16(a0, b1, acc01);
    acc10 = wmma_bf16(a1, b0, acc10);
    acc11 = wmma_bf16(a1, b1, acc11);
  }
  const int colL = lane & 15;
  const int rOff = (lane & 16) ? 8 : 0;
  v8f accs[2][2] = {{acc00, acc01}, {acc10, acc11}};
#pragma unroll
  for (int mi = 0; mi < 2; ++mi) {
#pragma unroll
    for (int ni = 0; ni < 2; ++ni) {
      const int col = nTile + ni * 16 + colL;
      const float bv = bias[col];
      const int rbase = mTile + mi * 16 + rOff;
#pragma unroll
      for (int v = 0; v < 8; ++v) {
        float val = (accs[mi][ni][v] + bv) * scale;
        size_t idx = (size_t)(rbase + v) * N + col;
        if (OUT_F32) ((float*)Cout)[idx] = val;
        else         ((__bf16*)Cout)[idx] = (__bf16)val;
      }
    }
  }
}

// ---------------------------------------------------------------------------
// V transpose: VT[b][h][d][m] = V[b*LKV+m][h*HDIM+d]  (coalesced writes)
// ---------------------------------------------------------------------------
__global__ void transpose_v(const __bf16* __restrict__ V, __bf16* __restrict__ VT) {
  size_t idx = (size_t)blockIdx.x * blockDim.x + threadIdx.x; // 2^22 total
  int m = (int)(idx & (LKV - 1));
  int d = (int)((idx >> 10) & (HDIM - 1));
  int h = (int)((idx >> 16) & (NHEADS - 1));
  int b = (int)(idx >> 20);
  VT[idx] = V[((size_t)(b * LKV + m)) * DIMK + h * HDIM + d];
}

// ---------------------------------------------------------------------------
// Flash attention. Block = 8 waves = 128 Q rows of one (b,h); each wave owns
// a 16-row Q tile. Key chunks of 32 are staged into LDS once per block with
// async-to-LDS loads (double buffered), shared by all 8 waves.
// Q is pre-scaled by 1/sqrt(HD) in its projection GEMM.
// ---------------------------------------------------------------------------
__global__ void flash_attn(const __bf16* __restrict__ Q, const __bf16* __restrict__ Kd,
                           const __bf16* __restrict__ VT, __bf16* __restrict__ O) {
  __shared__ __bf16 kTile[2][32 * 64];   // [key][d]   4 KB per buffer
  __shared__ __bf16 vTile[2][64 * 32];   // [d][key]   4 KB per buffer
  __shared__ __bf16 pTile[8][16 * 32];   // per-wave P bounce, 8 KB

  const int tid  = threadIdx.x;
  const int wave = tid >> 5;
  const int lane = tid & 31;
  const int b  = blockIdx.z;
  const int h  = blockIdx.y;
  const int m0 = blockIdx.x * 128 + wave * 16;

  const __bf16* qBase  = Q  + ((size_t)(b * LQ + m0)) * DIMK + h * HDIM;
  const __bf16* kBase  = Kd + ((size_t)b * LKV) * DIMK + h * HDIM;
  const __bf16* vtBase = VT + (((size_t)(b * NHEADS + h)) * HDIM) * LKV;
  __bf16* pmine = pTile[wave];

  // per-thread staging slots: one b128 of K tile + one b128 of VT tile
  const int kr = tid >> 3, kc = (tid & 7) * 8;   // K tile 32x64: 256 x 16B
  const int vr = tid >> 2, vc = (tid & 3) * 8;   // VT tile 64x32: 256 x 16B
  auto issue = [&](int key0, int buf) {
    async_load_b128((uint32_t)(uintptr_t)&kTile[buf][kr * 64 + kc],
                    kBase + (size_t)(key0 + kr) * DIMK + kc);
    async_load_b128((uint32_t)(uintptr_t)&vTile[buf][vr * 32 + vc],
                    vtBase + (size_t)vr * LKV + key0 + vc);
  };

  // Q fragments resident for the whole key loop
  v16bf qa0 = load_a_frag(qBase, DIMK);        // d = 0..31
  v16bf qa1 = load_a_frag(qBase + 32, DIMK);   // d = 32..63

  v8f accO[4];
#pragma unroll
  for (int t = 0; t < 4; ++t) accO[t] = {};
  v8f m8, l8;
#pragma unroll
  for (int v = 0; v < 8; ++v) { m8[v] = -1e30f; l8[v] = 0.0f; }

  const int col = lane & 15;
  const int hi8 = (lane & 16) ? 8 : 0;
  const int NCHUNK = LKV / 32;

  issue(0, 0);
  for (int j = 0; j < NCHUNK; ++j) {
    const int cur = j & 1;
    if (j + 1 < NCHUNK) {
      issue((j + 1) * 32, cur ^ 1);   // prefetch next chunk into other buffer
      wait_async_le2();               // our 2 ops for chunk j have landed
    } else {
      wait_async_0();
    }
    __syncthreads();                  // all waves' staging of chunk j visible

    const __bf16* kb = kTile[cur];
    const __bf16* vb = vTile[cur];

    // ---- scores S[16 x 32], 4 WMMA from LDS fragments ----
    v16bf kb00 = load_b_frag(kb, 64);
    v16bf kb01 = load_b_frag(kb + 32, 64);
    v16bf kb10 = load_b_frag(kb + 16 * 64, 64);
    v16bf kb11 = load_b_frag(kb + 16 * 64 + 32, 64);
    v8f s0 = {}, s1 = {};
    s0 = wmma_bf16(qa0, kb00, s0);
    s0 = wmma_bf16(qa1, kb01, s0);
    s1 = wmma_bf16(qa0, kb10, s1);
    s1 = wmma_bf16(qa1, kb11, s1);

    // ---- online softmax; rows live across 16-lane halves, DPP reductions ----
#pragma unroll
    for (int v = 0; v < 8; ++v) {
      float red = red_max16(fmaxf(s0[v], s1[v]));
      float mn  = fmaxf(m8[v], red);
      float al  = __expf(m8[v] - mn);
      float p0  = __expf(s0[v] - mn);
      float p1  = __expf(s1[v] - mn);
      float rs  = red_sum16(p0 + p1);
      l8[v] = l8[v] * al + rs;
      m8[v] = mn;
#pragma unroll
      for (int t = 0; t < 4; ++t) accO[t][v] *= al;
      pmine[(v + hi8) * 32 + col]      = (__bf16)p0;
      pmine[(v + hi8) * 32 + 16 + col] = (__bf16)p1;
    }
    // same-wave LDS ops are in-order; reload as A fragment (16x32)
    v16bf pa = load_a_frag(pmine, 32);

    // ---- O += P @ V, 4 WMMA from LDS VT tile ----
#pragma unroll
    for (int t = 0; t < 4; ++t) {
      v16bf vf = load_b_frag(vb + t * 16 * 32, 32);
      accO[t] = wmma_bf16(pa, vf, accO[t]);
    }
    __syncthreads();   // everyone done reading buf[cur] before it is reused
  }

  // ---- epilogue: divide by row sums, store bf16 attention output ----
  __bf16* oBase = O + ((size_t)(b * LQ + m0)) * DIMK + h * HDIM;
#pragma unroll
  for (int v = 0; v < 8; ++v) {
    float inv = 1.0f / l8[v];
#pragma unroll
    for (int t = 0; t < 4; ++t)
      oBase[(size_t)(v + hi8) * DIMK + t * 16 + col] = (__bf16)(accO[t][v] * inv);
  }
}

// ---------------------------------------------------------------------------
extern "C" void kernel_launch(void* const* d_in, const int* in_sizes, int n_in,
                              void* d_out, int out_size, void* d_ws, size_t ws_size,
                              hipStream_t stream) {
  const float* x_broad = (const float*)d_in[0];
  const float* x_low   = (const float*)d_in[1];
  const float* Wq = (const float*)d_in[2];
  const float* bq = (const float*)d_in[3];
  const float* Wk = (const float*)d_in[4];
  const float* bk = (const float*)d_in[5];
  const float* Wv = (const float*)d_in[6];
  const float* bv = (const float*)d_in[7];
  const float* Wo = (const float*)d_in[8];
  const float* bo = (const float*)d_in[9];

  const size_t NB = (size_t)BATCH * LQ * DIMK;   // 16.8M
  const size_t NL = (size_t)BATCH * LKV * DIMK;  //  4.2M
  const size_t NW = (size_t)DIMK * DIMK;         //  1.0M

  __bf16* ws = (__bf16*)d_ws;
  __bf16* xb  = ws; ws += NB;
  __bf16* xl  = ws; ws += NL;
  __bf16* wq  = ws; ws += NW;
  __bf16* wk  = ws; ws += NW;
  __bf16* wv  = ws; ws += NW;
  __bf16* wo  = ws; ws += NW;
  __bf16* Qb  = ws; ws += NB;
  __bf16* Kb  = ws; ws += NL;
  __bf16* Vb  = ws; ws += NL;
  __bf16* VTb = ws; ws += NL;
  __bf16* AOb = ws; ws += NB;

  auto cast = [&](const float* src, __bf16* dst, size_t n) {
    int n4 = (int)(n / 4);
    cast_f32_to_bf16<<<(n4 + 255) / 256, 256, 0, stream>>>(src, dst, n4);
  };
  cast(x_broad, xb, NB);
  cast(x_low,   xl, NL);
  cast(Wq, wq, NW);
  cast(Wk, wk, NW);
  cast(Wv, wv, NW);
  cast(Wo, wo, NW);

  const int Mq = BATCH * LQ;    // 16384
  const int Mk = BATCH * LKV;   //  4096
  dim3 blk(256);

  // projections (softmax SCALE folded into Q)
  gemm_xWt<0><<<dim3(Mq / 64, DIMK / 128), blk, 0, stream>>>(xb, wq, bq, Qb, Mq, DIMK, DIMK, QSCALE);
  gemm_xWt<0><<<dim3(Mk / 64, DIMK / 128), blk, 0, stream>>>(xl, wk, bk, Kb, Mk, DIMK, DIMK, 1.0f);
  gemm_xWt<0><<<dim3(Mk / 64, DIMK / 128), blk, 0, stream>>>(xl, wv, bv, Vb, Mk, DIMK, DIMK, 1.0f);

  transpose_v<<<(BATCH * NHEADS * HDIM * LKV) / 256, 256, 0, stream>>>(Vb, VTb);

  flash_attn<<<dim3(LQ / 128, NHEADS, BATCH), blk, 0, stream>>>(Qb, Kb, VTb, AOb);

  // output projection, fp32 directly into d_out
  gemm_xWt<1><<<dim3(Mq / 64, DIMK / 128), blk, 0, stream>>>(AOb, wo, bo, (float*)d_out, Mq, DIMK, DIMK, 1.0f);
}
